// W8A8Linear_50405736186186
// MI455X (gfx1250) — 8-bit matrix operand order repaired and checked by reading, compiled, not run on hardware
//
#include <hip/hip_runtime.h>
#include <stdint.h>

typedef __attribute__((ext_vector_type(8))) int v8i;

#define QMAXF 127.0f

__global__ void __launch_bounds__(256)
quantize_rows_kernel(const float* __restrict__ x,
                     int8_t* __restrict__ xq,
                     float* __restrict__ xscale,
                     int K  ) {
    const int row = blockIdx.x;
    const int t   = threadIdx.x;
    const float* xr = x + (size_t)row * K;

    float4 v[4];
    float m = 0.0f;
#pragma unroll
    for (int j = 0; j < 4; ++j) {
        v[j] = reinterpret_cast<const float4*>(xr)[t + j * 256];
        m = fmaxf(m, fmaxf(fmaxf(fabsf(v[j].x), fabsf(v[j].y)),
                           fmaxf(fabsf(v[j].z), fabsf(v[j].w))));
    }

    __shared__ float red[256];
    red[t] = m;
    __syncthreads();
#pragma unroll
    for (int s = 128; s > 0; s >>= 1) {
        if (t < s) red[t] = fmaxf(red[t], red[t + s]);
        __syncthreads();
    }
    const float amax = red[0];
    const float sc   = fmaxf(amax, 1e-8f) * (1.0f / QMAXF);
    const float inv  = 1.0f / sc;
    if (t == 0) xscale[row] = sc;

    uint32_t* dst = reinterpret_cast<uint32_t*>(xq + (size_t)row * K);
#pragma unroll
    for (int j = 0; j < 4; ++j) {
        int q0 = (int)fminf(fmaxf(rintf(v[j].x * inv), -QMAXF), QMAXF);
        int q1 = (int)fminf(fmaxf(rintf(v[j].y * inv), -QMAXF), QMAXF);
        int q2 = (int)fminf(fmaxf(rintf(v[j].z * inv), -QMAXF), QMAXF);
        int q3 = (int)fminf(fmaxf(rintf(v[j].w * inv), -QMAXF), QMAXF);
        uint32_t pk = (uint32_t)(q0 & 0xff)         |
                      ((uint32_t)(q1 & 0xff) << 8)  |
                      ((uint32_t)(q2 & 0xff) << 16) |
                      ((uint32_t)(q3 & 0xff) << 24);
        dst[t + j * 256] = pk;
    }
}

__global__ void __launch_bounds__(256)
w8a8_gemm_wmma_kernel(const int8_t* __restrict__ xq,
                      const int8_t* __restrict__ wgt,
                      const float* __restrict__ xs,
                      const float* __restrict__ wsc,
                      const float* __restrict__ bias,
                      float* __restrict__ out,
                      int N, int K) {
    const int tid    = threadIdx.x;
    const int wave   = tid >> 5;
    const int lane   = tid & 31;
    const int lane15 = lane & 15;
    const int khalf  = lane >> 4;

    const int waveM = wave >> 2;
    const int waveN = wave & 3;
    const int m0 = blockIdx.y * 128 + waveM * 64;
    const int n0 = blockIdx.x * 256 + waveN * 64;

    v8i acc[4][4];
#pragma unroll
    for (int mt = 0; mt < 4; ++mt)
#pragma unroll
        for (int nt = 0; nt < 4; ++nt)
            acc[mt][nt] = (v8i)(0);

    const int8_t* arow[4];
#pragma unroll
    for (int mt = 0; mt < 4; ++mt)
        arow[mt] = xq + (size_t)(m0 + mt * 16 + lane15) * K;

    const int8_t* brow[4];
#pragma unroll
    for (int nt = 0; nt < 4; ++nt)
        brow[nt] = wgt + (size_t)(n0 + nt * 16 + lane15) * K;

    for (int kk = 0; kk < K; kk += 64) {
#pragma unroll
        for (int nt = 0; nt < 4; ++nt)
            __builtin_prefetch(brow[nt] + kk + 64 + khalf * 16, 0, 3);
#pragma unroll
        for (int mt = 0; mt < 4; ++mt)
            __builtin_prefetch(arow[mt] + kk + 64 + khalf * 8, 0, 3);

        v8i afrag[4];
#pragma unroll
        for (int mt = 0; mt < 4; ++mt) {
            const int8_t* p = arow[mt] + kk + khalf * 16;
            uint4 d0 = *reinterpret_cast<const uint4*>(p);
            uint4 d1 = *reinterpret_cast<const uint4*>(p + 32);
            v8i a;
            a[0] = (int)d0.x; a[1] = (int)d0.y;
            a[2] = (int)d0.z; a[3] = (int)d0.w;
            a[4] = (int)d1.x; a[5] = (int)d1.y;
            a[6] = (int)d1.z; a[7] = (int)d1.w;
            afrag[mt] = a;
        }

#pragma unroll
        for (int nt = 0; nt < 4; ++nt) {
            const int8_t* p = brow[nt] + kk + khalf * 16;
            uint4 q0 = *reinterpret_cast<const uint4*>(p);
            uint4 q1 = *reinterpret_cast<const uint4*>(p + 32);
            v8i b;
            b[0] = (int)q0.x; b[1] = (int)q0.y; b[2] = (int)q0.z; b[3] = (int)q0.w;
            b[4] = (int)q1.x; b[5] = (int)q1.y; b[6] = (int)q1.z; b[7] = (int)q1.w;
#pragma unroll
            for (int mt = 0; mt < 4; ++mt)
                acc[mt][nt] = __builtin_amdgcn_wmma_i32_16x16x64_iu8(
                    true, afrag[mt], true, b, acc[mt][nt],
                    false, false);
        }
    }

    float wsc_l[4], bias_l[4];
#pragma unroll
    for (int nt = 0; nt < 4; ++nt) {
        const int n = n0 + nt * 16 + lane15;
        wsc_l[nt]  = wsc[n];
        bias_l[nt] = bias[n];
    }

#pragma unroll
    for (int mt = 0; mt < 4; ++mt) {
#pragma unroll
        for (int g = 0; g < 8; ++g) {
            const int row = m0 + mt * 16 + g + khalf * 8;
            const float s = xs[row];
            float* orow = out + (size_t)row * N;
#pragma unroll
            for (int nt = 0; nt < 4; ++nt) {
                const int n = n0 + nt * 16 + lane15;
                orow[n] = (float)acc[mt][nt][g] * s * wsc_l[nt] + bias_l[nt];
            }
        }
    }
}

extern "C" void kernel_launch(void* const* d_in, const int* in_sizes, int n_in,
                              void* d_out, int out_size, void* d_ws, size_t ws_size,
                              hipStream_t stream) {
    const float*  x    = (const float*)d_in[0];
    const int8_t* wgt  = (const int8_t*)d_in[1];
    const float*  wsc  = (const float*)d_in[2];
    const float*  bias = (const float*)d_in[3];
    float* out = (float*)d_out;

    const int OUT = in_sizes[2];
    const int K   = in_sizes[1] / OUT;
    const int M   = in_sizes[0] / K;

    int8_t* xq     = (int8_t*)d_ws;
    float*  xscale = (float*)((char*)d_ws + (size_t)M * K);

    quantize_rows_kernel<<<dim3(M), dim3(256), 0, stream>>>(x, xq, xscale, K);

    dim3 grid(OUT / 256, M / 128);
    w8a8_gemm_wmma_kernel<<<grid, dim3(256), 0, stream>>>(
        xq, wgt, xscale, wsc, bias, out, OUT, K);
}
